// TreeLSTM_53523882443642
// MI455X (gfx1250) — matrix operand order repaired and checked by reading, not compiled, not run on hardware
//
#include <hip/hip_runtime.h>
#include <hip/hip_bf16.h>
#include <math.h>

typedef __attribute__((ext_vector_type(16))) _Float16 v16h;
typedef __attribute__((ext_vector_type(8)))  _Float16 v8h;
typedef __attribute__((ext_vector_type(8)))  float    v8f;

#define HSZ   256
#define KDIM  512            // 2*H (GEMM reduction dim)
#define DEPTH 17
#define NTOT  ((1 << DEPTH) - 1)
#define EPSV  1e-12f

__device__ __forceinline__ float sigm(float x) { return 1.0f / (1.0f + __expf(-x)); }

// ---------------------------------------------------------------------------
// Pack fp32 row-major weights W[K][Ncols] into WMMA B-fragment order (fp16).
// Packed index: p = ((kk*nTiles + tc)*32 + lane)*16 + t
//   K = kk*32 + (lane/16)*16 + t         (contiguous k order, same as the A fragment)
//   N = tc*16 + (lane%16)
// Each lane's 16 halves are contiguous (32B) -> wave reads contiguous 1KB.
// ---------------------------------------------------------------------------
__global__ void pack_weights(const float* __restrict__ W, _Float16* __restrict__ P,
                             int Ncols, int total) {
  int p = blockIdx.x * blockDim.x + threadIdx.x;
  if (p >= total) return;
  int nTiles = Ncols >> 4;
  int t    = p & 15;
  int lane = (p >> 4) & 31;
  int kt   = p >> 9;                 // kk*nTiles + tc
  int tc   = kt % nTiles;
  int kk   = kt / nTiles;
  int K = kk * 32 + ((lane >> 4) << 4) + t;
  int Nc = tc * 16 + (lane & 15);
  P[p] = (_Float16)W[(long)K * Ncols + Nc];
}

// ---------------------------------------------------------------------------
// Leaves: apply_node(iou + b_iou, c0)  -> h (d_out), c (workspace)
// ---------------------------------------------------------------------------
__global__ void leaf_kernel(const float* __restrict__ iou, const float* __restrict__ c0,
                            const float* __restrict__ b_iou,
                            float* __restrict__ hbuf, float* __restrict__ cbuf,
                            int leafStart) {
  long idx = (long)blockIdx.x * blockDim.x + threadIdx.x;   // leaves * 256
  int  col  = (int)(idx & (HSZ - 1));
  long node = leafStart + (idx >> 8);
  if (node >= NTOT) return;
  const float* r = iou + node * (3 * HSZ);
  float i = r[col]           + b_iou[col];
  float o = r[HSZ + col]     + b_iou[HSZ + col];
  float u = r[2 * HSZ + col] + b_iou[2 * HSZ + col];
  float c = sigm(i) * tanhf(u) + c0[node * HSZ + col];
  float h = sigm(o) * tanhf(c);
  hbuf[node * HSZ + col] = h;
  cbuf[node * HSZ + col] = c;
}

// ---------------------------------------------------------------------------
// One tree level: 16 nodes per 512-thread block (16 waves).
// Children of the node tile are contiguous in hbuf: async-stage the 32x256 f32
// block into LDS once (ASYNCcnt path); the staged block IS h_cat row-major.
// Wave w = column group; 5 accumulators (f0,f1 @ F cols {w,w+16};
// i,o,u @ U cols {w,w+16,w+32}) -> gates element-aligned per lane.
// ---------------------------------------------------------------------------
__launch_bounds__(512)
__global__ void level_kernel(const float* __restrict__ iou,
                             const float* __restrict__ b_f,
                             const float* __restrict__ b_iou,
                             const float* __restrict__ scale_iou,
                             const float* __restrict__ scale_c,
                             const _Float16* __restrict__ WfP,   // packed 512x512
                             const _Float16* __restrict__ WiP,   // packed 512x768
                             float* __restrict__ hbuf,
                             float* __restrict__ cbuf,
                             int s0, int e0) {
  // 32KB: children h (16 rows x 512 = h_cat row-major); reused as c_sum stage.
  __shared__ __align__(16) float    Hstage[16 * KDIM];
  // 16KB: fp16 A matrix for WMMA.
  __shared__ __align__(16) _Float16 Alds[16 * KDIM];
  __shared__ float siou_s[16];   // ||iou|| * scale_iou / max(||h_cat||, eps)
  __shared__ float cn0_s[16];    // ||c_child0||
  __shared__ float csn_s[16];    // ||c_sum||

  const int tid  = threadIdx.x;          // 0..511
  const int wid  = tid >> 5;             // 0..15 (wave = column group)
  const int lane = tid & 31;
  const int nodeBase = s0 + blockIdx.x * 16;
  const int validNodes = (e0 - nodeBase) < 16 ? (e0 - nodeBase) : 16;
  const int validChildRows = 2 * validNodes;   // child rows present in hbuf

  // ---- Phase 1a: async-stage contiguous children-h block into LDS --------
  {
    const float* src = hbuf + (long)(2 * nodeBase + 1) * HSZ;  // 32 x 256 f32
    for (int j = tid; j < (16 * KDIM) / 4; j += 512) {         // b128 chunks
      int childRow = j >> 6;                                   // (j*4)/256
      unsigned ldsOff = (unsigned)(uintptr_t)(&Hstage[0]) + (unsigned)j * 16u;
      if (childRow < validChildRows) {
        unsigned long long ga = (unsigned long long)(uintptr_t)(src + j * 4);
        asm volatile("global_load_async_to_lds_b128 %0, %1, off"
                     :: "v"(ldsOff), "v"(ga) : "memory");
      } else {
        *reinterpret_cast<float4*>(&Hstage[j * 4]) = make_float4(0.f, 0.f, 0.f, 0.f);
      }
    }
    asm volatile("s_wait_asynccnt 0x0" ::: "memory");
  }
  __syncthreads();

  // ---- Phase 1b: deterministic row norms (wave wid owns node row wid) ----
  {
    int node = nodeBase + wid;
    float sh = 0.f, si = 0.f, sc0 = 0.f;
    if (node < e0) {
      for (int j = lane; j < KDIM; j += 32) { float v = Hstage[wid * KDIM + j]; sh += v * v; }
      long ib = (long)node * (3 * HSZ);
      for (int j = lane; j < 3 * HSZ; j += 32) { float v = iou[ib + j]; si += v * v; }
      long ch0 = (long)(2 * node + 1) * HSZ;
      for (int j = lane; j < HSZ; j += 32)     { float v = cbuf[ch0 + j]; sc0 += v * v; }
    }
    for (int m = 16; m >= 1; m >>= 1) {
      sh  += __shfl_xor(sh,  m, 32);
      si  += __shfl_xor(si,  m, 32);
      sc0 += __shfl_xor(sc0, m, 32);
    }
    if (lane == 0) {
      siou_s[wid] = sqrtf(si) * scale_iou[0] / fmaxf(sqrtf(sh), EPSV);
      cn0_s[wid]  = sqrtf(sc0);
    }
  }

  // ---- Phase 1c: fp16 A matrix (stage block is already h_cat row-major) --
  for (int idx = tid; idx < 16 * KDIM; idx += 512)
    Alds[idx] = (_Float16)Hstage[idx];
  __syncthreads();

  // ---- Phase 2: WMMA GEMMs ----------------------------------------------
  const int w    = wid;          // column group
  const int lrow = lane & 15;
  const int lhi  = lane >> 4;

  v8f accF0, accF1, accI, accO, accU;
#pragma unroll
  for (int t = 0; t < 8; ++t) { accF0[t] = 0.f; accF1[t] = 0.f; accI[t] = 0.f; accO[t] = 0.f; accU[t] = 0.f; }

  const int arowBase = lrow * KDIM;
  for (int kk = 0; kk < KDIM / 32; ++kk) {
    const int k0 = kk * 32;
    // A fragment (contiguous order, matches packed B): halves 0..15 -> K=k0+base+t,
    // base = (lane/16)*16
    v16h a;
    {
      int aoff = arowBase + k0 + lhi * 16;
      v8h lo = *(const v8h*)&Alds[aoff];
      v8h hi = *(const v8h*)&Alds[aoff + 8];
#pragma unroll
      for (int t = 0; t < 8; ++t) { a[t] = lo[t]; a[8 + t] = hi[t]; }
    }
    // B fragments: 16 contiguous halves per lane (32B), 1KB/wave contiguous
    v16h bF0 = *(const v16h*)(WfP + ((((long)kk * 32 + (w))      * 32 + lane) << 4));
    v16h bF1 = *(const v16h*)(WfP + ((((long)kk * 32 + (w + 16)) * 32 + lane) << 4));
    v16h bI  = *(const v16h*)(WiP + ((((long)kk * 48 + (w))      * 32 + lane) << 4));
    v16h bO  = *(const v16h*)(WiP + ((((long)kk * 48 + (w + 16)) * 32 + lane) << 4));
    v16h bU  = *(const v16h*)(WiP + ((((long)kk * 48 + (w + 32)) * 32 + lane) << 4));
    accF0 = __builtin_amdgcn_wmma_f32_16x16x32_f16(false, a, false, bF0, (short)0, accF0, false, false);
    accF1 = __builtin_amdgcn_wmma_f32_16x16x32_f16(false, a, false, bF1, (short)0, accF1, false, false);
    accI  = __builtin_amdgcn_wmma_f32_16x16x32_f16(false, a, false, bI,  (short)0, accI,  false, false);
    accO  = __builtin_amdgcn_wmma_f32_16x16x32_f16(false, a, false, bO,  (short)0, accO,  false, false);
    accU  = __builtin_amdgcn_wmma_f32_16x16x32_f16(false, a, false, bU,  (short)0, accU,  false, false);
  }

  // ---- Phase 3a: forget gates, c_sum; stage c_sum (reuse Hstage) ---------
  const int col = w * 16 + lrow;      // C layout: N = lane%16
  float csum[8];
  int   nodev[8];
  bool  okv[8];
#pragma unroll
  for (int v = 0; v < 8; ++v) {
    int mLocal = v + 8 * lhi;         // C layout: M = v + 8*(lane/16)
    int node = nodeBase + mLocal;
    bool ok = node < e0;
    float f0 = sigm(accF0[v] + b_f[col]);
    float f1 = sigm(accF1[v] + b_f[HSZ + col]);
    float c0v = ok ? cbuf[(long)(2 * node + 1) * HSZ + col] : 0.f;
    float c1v = ok ? cbuf[(long)(2 * node + 2) * HSZ + col] : 0.f;
    float cs = f0 * c0v + f1 * c1v;
    csum[v] = cs; nodev[v] = node; okv[v] = ok;
    Hstage[mLocal * HSZ + col] = cs;  // first 16KB of stage, no live h left
  }
  __syncthreads();

  // ---- Phase 3b: ||c_sum|| per row (deterministic wave reduce) -----------
  {
    float s = 0.f;
    for (int j = lane; j < HSZ; j += 32) { float v = Hstage[wid * HSZ + j]; s += v * v; }
    for (int m = 16; m >= 1; m >>= 1) s += __shfl_xor(s, m, 32);
    if (lane == 0) csn_s[wid] = sqrtf(s);
  }
  __syncthreads();

  // ---- Phase 3c: msg_norm(c), gates, write h/c ---------------------------
  const float sc = scale_c[0];
#pragma unroll
  for (int v = 0; v < 8; ++v) {
    int mLocal = v + 8 * lhi;
    float cin = csum[v] / fmaxf(csn_s[mLocal], EPSV) * cn0_s[mLocal] * sc;
    float si  = siou_s[mLocal];                 // msg_norm row-scale, folded out of GEMM
    float i = accI[v] * si + b_iou[col];
    float o = accO[v] * si + b_iou[HSZ + col];
    float u = accU[v] * si + b_iou[2 * HSZ + col];
    float cnew = sigm(i) * tanhf(u) + cin;
    float hnew = sigm(o) * tanhf(cnew);
    if (okv[v]) {
      hbuf[(long)nodev[v] * HSZ + col] = hnew;
      cbuf[(long)nodev[v] * HSZ + col] = cnew;
    }
  }
}

// ---------------------------------------------------------------------------
extern "C" void kernel_launch(void* const* d_in, const int* in_sizes, int n_in,
                              void* d_out, int out_size, void* d_ws, size_t ws_size,
                              hipStream_t stream) {
  (void)in_sizes; (void)n_in; (void)out_size; (void)ws_size;
  const float* iou   = (const float*)d_in[0];
  // d_in[1] = h0 (unused by the reference beyond shape)
  const float* c0    = (const float*)d_in[2];
  const float* Wiou  = (const float*)d_in[3];
  const float* Wf    = (const float*)d_in[4];
  const float* b_f   = (const float*)d_in[5];
  const float* b_iou = (const float*)d_in[6];
  const float* s_iou = (const float*)d_in[7];
  const float* s_c   = (const float*)d_in[8];
  float* hbuf = (float*)d_out;

  // Workspace: c state (N*256 f32) + packed fp16 weights
  float*    cbuf = (float*)d_ws;
  size_t    coff = (size_t)NTOT * HSZ * sizeof(float);          // 134,216,704 B
  _Float16* WfP  = (_Float16*)((char*)d_ws + coff);             // 512*512 halves
  _Float16* WiP  = WfP + 512 * 512;                             // 512*768 halves

  pack_weights<<<(512 * 512 + 255) / 256, 256, 0, stream>>>(Wf,   WfP, 512, 512 * 512);
  pack_weights<<<(512 * 768 + 255) / 256, 256, 0, stream>>>(Wiou, WiP, 768, 512 * 768);

  const int leafStart = (1 << (DEPTH - 1)) - 1;                 // 65535
  const long leafElems = (long)(1 << (DEPTH - 1)) * HSZ;        // 65536*256
  leaf_kernel<<<(int)(leafElems / 256), 256, 0, stream>>>(iou, c0, b_iou, hbuf, cbuf, leafStart);

  for (int lvl = DEPTH - 2; lvl >= 0; --lvl) {                  // bottom-up levels
    int s0 = (1 << lvl) - 1;
    int n  = 1 << lvl;
    int e0 = s0 + n;
    int blocks = (n + 15) / 16;
    level_kernel<<<blocks, 512, 0, stream>>>(iou, b_f, b_iou, s_iou, s_c,
                                             WfP, WiP, hbuf, cbuf, s0, e0);
  }
}
